// PhonemeAutoencoder_18176301597594
// MI455X (gfx1250) — compile-verified, matrix operand order checked by reading, not run on hardware
//
#include <hip/hip_runtime.h>
#include <math.h>

// ---------------- CDNA5 WMMA types ----------------
typedef __attribute__((ext_vector_type(16))) __bf16 v16bf;
typedef __attribute__((ext_vector_type(8)))  float  v8f;

#define WMMA_BF16(a, b, c) \
    __builtin_amdgcn_wmma_f32_16x16x32_bf16(false, (a), false, (b), (short)0, (c), false, false)

// Problem constants (match reference)
#define CB   16
#define CT   128
#define CTP  32
#define CV   256
#define CE   128
#define CH   256
#define CBT  (CB * CT)      // 2048
#define CNG  (4 * CH)       // 1024

// ---------------- helpers ----------------
__device__ __forceinline__ __bf16 f2bf(float f) {
    union { float f; unsigned u; } a; a.f = f;
    unsigned r = a.u + 0x7FFFu + ((a.u >> 16) & 1u);   // round-to-nearest-even
    union { unsigned short s; __bf16 b; } o; o.s = (unsigned short)(r >> 16);
    return o.b;
}

__device__ __forceinline__ float sigmf(float x) { return 1.0f / (1.0f + __expf(-x)); }

// Build a 16x32 bf16 A-fragment from a bf16 row (ISA 16-bit A layout):
// lane = M (mod 16), kh = lane/16 selects K-half; element e: g=e>>3, w=e&7,
// K = g*16 + kh*8 + w  -> two contiguous 8-element (16B) groups per lane.
__device__ __forceinline__ v16bf loadA_bf16(const __bf16* __restrict__ row, int kBase, int kh) {
    v16bf a;
    const __bf16* p0 = row + kBase + kh * 8;
    const __bf16* p1 = p0 + 16;
#pragma unroll
    for (int w = 0; w < 8; ++w) { a[w] = p0[w]; a[8 + w] = p1[w]; }
    return a;
}

// ---------------- setup kernels ----------------
__global__ void convert_emb_kernel(const float* __restrict__ emb, __bf16* __restrict__ embp) {
    int i = blockIdx.x * blockDim.x + threadIdx.x;
    if (i < CV * CE) embp[i] = f2bf(emb[i]);
}

__global__ void combine_bias_kernel(const float* __restrict__ ebih, const float* __restrict__ ebhh,
                                    const float* __restrict__ dbih, const float* __restrict__ dbhh,
                                    float* __restrict__ enc_b, float* __restrict__ dec_b) {
    int i = blockIdx.x * blockDim.x + threadIdx.x;
    if (i < CNG) { enc_b[i] = ebih[i] + ebhh[i]; dec_b[i] = dbih[i] + dbhh[i]; }
}

// Pack [Wih (1024x128) ; Whh (1024x256)] into WMMA B-fragment tiles:
// tile (nt, kt) -> 32 lanes x 16 bf16 contiguous; tiles ordered nt-major, 12 kt per nt.
__global__ void pack_gates_kernel(const float* __restrict__ Wih, const float* __restrict__ Whh,
                                  __bf16* __restrict__ dst) {
    int tid = blockIdx.x * blockDim.x + threadIdx.x;     // (nt*12+kt)*32 + lane
    if (tid >= 64 * 12 * 32) return;
    int lane = tid & 31;
    int tile = tid >> 5;
    int kt = tile % 12, nt = tile / 12;
    int n  = nt * 16 + (lane & 15);
    int kh = lane >> 4;
    __bf16* o = dst + (size_t)tid * 16;
#pragma unroll
    for (int e = 0; e < 16; ++e) {
        int k = kt * 32 + (e >> 3) * 16 + kh * 8 + (e & 7);
        float v = (kt < 4) ? Wih[(size_t)n * CE + k] : Whh[(size_t)n * CH + (k - CE)];
        o[e] = f2bf(v);
    }
}

__global__ void pack_fc_kernel(const float* __restrict__ W, __bf16* __restrict__ dst) {
    int tid = blockIdx.x * blockDim.x + threadIdx.x;     // (nt*8+kt)*32 + lane
    if (tid >= 16 * 8 * 32) return;
    int lane = tid & 31;
    int tile = tid >> 5;
    int kt = tile % 8, nt = tile / 8;
    int n  = nt * 16 + (lane & 15);
    int kh = lane >> 4;
    __bf16* o = dst + (size_t)tid * 16;
#pragma unroll
    for (int e = 0; e < 16; ++e) {
        int k = kt * 32 + (e >> 3) * 16 + kh * 8 + (e & 7);
        o[e] = f2bf(W[(size_t)n * CH + k]);
    }
}

__global__ void init_state_kernel(unsigned short* __restrict__ h0, float* __restrict__ c0,
                                  float* __restrict__ cd, int* __restrict__ tok,
                                  const int* __restrict__ sos) {
    int i = blockIdx.x * blockDim.x + threadIdx.x;
    if (i < CBT * CH) { h0[i] = 0; c0[i] = 0.0f; cd[i] = 0.0f; }
    if (i < CBT) tok[i] = *sos;
}

// ---------------- fused LSTM step (gates GEMM + cell, WMMA bf16) ----------------
// grid (BT/16, 4), block 128 (4 waves). Wave w handles hidden-unit tile
// ntH = blockIdx.y*4+w and computes the 4 gate tiles {ntH, ntH+16, ntH+32, ntH+48}
// with one shared A fragment -> i/f/g/o for a (row, unit) land in the same lane,
// so the whole LSTM cell runs in registers; h is produced directly in bf16.
__global__ __launch_bounds__(128) void lstm_step_fused(
    const int* __restrict__ gidx, int gstride, int goff,   // per-row token index
    const __bf16* __restrict__ embp,                       // V x E bf16
    const __bf16* __restrict__ hin,                        // BT x H bf16
    const __bf16* __restrict__ Wp,                         // packed gates weights
    const float* __restrict__ bias,                        // 4H combined bias
    __bf16* __restrict__ hout,                             // BT x H bf16
    float* __restrict__ cst,                               // BT x H f32
    __bf16* __restrict__ enc_last,                         // BT x H bf16 (nullable)
    const int* __restrict__ lens, int t) {
    const int KT = 12;
    int lane = threadIdx.x & 31;
    int wave = threadIdx.x >> 5;
    int mTile = blockIdx.x;
    int ntH = blockIdx.y * 4 + wave;                       // 0..15
    int m = lane & 15, kh = lane >> 4;
    int row = mTile * 16 + m;

    v8f ai = {}, af = {}, ag = {}, ao = {};
    const size_t tb_i = (size_t)(ntH)      * KT * 512;
    const size_t tb_f = (size_t)(ntH + 16) * KT * 512;
    const size_t tb_g = (size_t)(ntH + 32) * KT * 512;
    const size_t tb_o = (size_t)(ntH + 48) * KT * 512;
    const size_t lb = (size_t)lane * 16;

    // x-part: gathered embedding rows (bf16, L2-resident)
    int idx = gidx[row * gstride + goff];
    const __bf16* erow = embp + (size_t)idx * CE;
#pragma unroll
    for (int kt = 0; kt < 4; ++kt) {
        v16bf a  = loadA_bf16(erow, kt * 32, kh);
        v16bf bi = *(const v16bf*)(Wp + tb_i + (size_t)kt * 512 + lb);
        v16bf bf = *(const v16bf*)(Wp + tb_f + (size_t)kt * 512 + lb);
        v16bf bg = *(const v16bf*)(Wp + tb_g + (size_t)kt * 512 + lb);
        v16bf bo = *(const v16bf*)(Wp + tb_o + (size_t)kt * 512 + lb);
        ai = WMMA_BF16(a, bi, ai);
        af = WMMA_BF16(a, bf, af);
        ag = WMMA_BF16(a, bg, ag);
        ao = WMMA_BF16(a, bo, ao);
    }
    // h-part: bf16 state, straight b128 fragment loads (no conversion)
    const __bf16* hrow = hin + (size_t)row * CH;
#pragma unroll
    for (int kt = 0; kt < 8; ++kt) {
        v16bf a  = loadA_bf16(hrow, kt * 32, kh);
        v16bf bi = *(const v16bf*)(Wp + tb_i + (size_t)(4 + kt) * 512 + lb);
        v16bf bf = *(const v16bf*)(Wp + tb_f + (size_t)(4 + kt) * 512 + lb);
        v16bf bg = *(const v16bf*)(Wp + tb_g + (size_t)(4 + kt) * 512 + lb);
        v16bf bo = *(const v16bf*)(Wp + tb_o + (size_t)(4 + kt) * 512 + lb);
        ai = WMMA_BF16(a, bi, ai);
        af = WMMA_BF16(a, bf, af);
        ag = WMMA_BF16(a, bg, ag);
        ao = WMMA_BF16(a, bo, ao);
    }

    // fused cell epilogue: C layout -> lane holds n=lane&15, VGPR r holds m = kh*8+r
    int nn = ntH * 16 + m;
    float bI = bias[nn], bF = bias[CH + nn], bG = bias[2 * CH + nn], bO = bias[3 * CH + nn];
#pragma unroll
    for (int r = 0; r < 8; ++r) {
        int mm = mTile * 16 + kh * 8 + r;
        size_t off = (size_t)mm * CH + nn;
        float iv = sigmf(ai[r] + bI);
        float fv = sigmf(af[r] + bF);
        float gv = tanhf(ag[r] + bG);
        float ov = sigmf(ao[r] + bO);
        float cv = fv * cst[off] + iv * gv;
        float hv = ov * tanhf(cv);
        cst[off]  = cv;
        __bf16 hb = f2bf(hv);
        hout[off] = hb;
        if (enc_last != nullptr && (lens[mm] - 1 == t)) enc_last[off] = hb;
    }
}

// ---------------- FC GEMM: scores = h @ fc_W^T + fc_b, written into d_out ----------------
__global__ __launch_bounds__(128) void fc_gemm(
    const __bf16* __restrict__ A,         // BT x 256 bf16 (h)
    const __bf16* __restrict__ Wp,        // packed 16 nTiles x 8 kTiles
    const float* __restrict__ bias,       // 256
    float* __restrict__ C, long ldC) {
    int lane = threadIdx.x & 31;
    int wave = threadIdx.x >> 5;
    int mTile = blockIdx.x;
    int nTile = blockIdx.y * 4 + wave;    // 0..15
    int m = lane & 15, kh = lane >> 4;
    int row = mTile * 16 + m;
    const __bf16* arow = A + (size_t)row * CH;
    const size_t tb = (size_t)nTile * 8 * 512;
    const size_t lb = (size_t)lane * 16;
    v8f acc = {};
#pragma unroll
    for (int kt = 0; kt < 8; ++kt) {
        v16bf a = loadA_bf16(arow, kt * 32, kh);
        v16bf b = *(const v16bf*)(Wp + tb + (size_t)kt * 512 + lb);
        acc = WMMA_BF16(a, b, acc);
    }
    int nn = nTile * 16 + m;
    float bv = bias[nn];
#pragma unroll
    for (int r = 0; r < 8; ++r) {
        int mm = mTile * 16 + kh * 8 + r;
        C[(size_t)mm * ldC + nn] = acc[r] + bv;
    }
}

// ---------------- greedy argmax over V=256 -> next token (one wave32 per row) ----------------
__global__ __launch_bounds__(256) void argmax_kernel(const float* __restrict__ scores, long ldS,
                                                     int* __restrict__ tok) {
    int wid  = (blockIdx.x * blockDim.x + threadIdx.x) >> 5;   // row
    int lane = threadIdx.x & 31;
    const float* s = scores + (size_t)wid * ldS;
    float best = -INFINITY;
    int bi = 0;
#pragma unroll
    for (int j = 0; j < CV / 32; ++j) {
        int v = lane + j * 32;                 // coalesced strided read
        float x = s[v];
        if (x > best || (x == best && v < bi)) { best = x; bi = v; }
    }
#pragma unroll
    for (int off = 16; off > 0; off >>= 1) {   // wave32 reduction, first-occurrence tie-break
        float ob = __shfl_down(best, off, 32);
        int   oi = __shfl_down(bi, off, 32);
        if (ob > best || (ob == best && oi < bi)) { best = ob; bi = oi; }
    }
    if (lane == 0) tok[wid] = bi;
}

// ---------------- host launch ----------------
extern "C" void kernel_launch(void* const* d_in, const int* in_sizes, int n_in,
                              void* d_out, int out_size, void* d_ws, size_t ws_size,
                              hipStream_t stream) {
    (void)in_sizes; (void)n_in; (void)out_size; (void)ws_size;
    const int* phonemes = (const int*)d_in[0];
    const int* lens     = (const int*)d_in[1];
    const float* emb    = (const float*)d_in[2];
    const float* eWih   = (const float*)d_in[3];
    const float* eWhh   = (const float*)d_in[4];
    const float* ebih   = (const float*)d_in[5];
    const float* ebhh   = (const float*)d_in[6];
    const float* dWih   = (const float*)d_in[7];
    const float* dWhh   = (const float*)d_in[8];
    const float* dbih   = (const float*)d_in[9];
    const float* dbhh   = (const float*)d_in[10];
    const float* fcW    = (const float*)d_in[11];
    const float* fcb    = (const float*)d_in[12];
    const int* sos      = (const int*)d_in[13];
    float* out = (float*)d_out;

    // workspace bump allocator (256B aligned)
    char* ws = (char*)d_ws;
    auto alloc = [&](size_t bytes) -> void* {
        void* p = (void*)ws;
        ws += (bytes + 255) & ~(size_t)255;
        return p;
    };
    __bf16* embp  = (__bf16*)alloc((size_t)CV * CE * 2);
    __bf16* encWp = (__bf16*)alloc((size_t)64 * 12 * 512 * 2);
    __bf16* decWp = (__bf16*)alloc((size_t)64 * 12 * 512 * 2);
    __bf16* fcWp  = (__bf16*)alloc((size_t)16 * 8 * 512 * 2);
    float* enc_b    = (float*)alloc(CNG * 4);
    float* dec_b    = (float*)alloc(CNG * 4);
    __bf16* hb0     = (__bf16*)alloc((size_t)CBT * CH * 2);   // encoder h ping
    __bf16* hb1     = (__bf16*)alloc((size_t)CBT * CH * 2);   // encoder h pong
    float*  c_enc   = (float*)alloc((size_t)CBT * CH * 4);
    __bf16* enc_last= (__bf16*)alloc((size_t)CBT * CH * 2);   // also decoder h ping
    __bf16* hd1     = (__bf16*)alloc((size_t)CBT * CH * 2);   // decoder h pong
    float*  c_dec   = (float*)alloc((size_t)CBT * CH * 4);
    int*    tok     = (int*)alloc(CBT * 4);

    // ---- setup ----
    convert_emb_kernel<<<(CV * CE + 255) / 256, 256, 0, stream>>>(emb, embp);
    combine_bias_kernel<<<(CNG + 255) / 256, 256, 0, stream>>>(ebih, ebhh, dbih, dbhh, enc_b, dec_b);
    pack_gates_kernel<<<(64 * 12 * 32 + 255) / 256, 256, 0, stream>>>(eWih, eWhh, encWp);
    pack_gates_kernel<<<(64 * 12 * 32 + 255) / 256, 256, 0, stream>>>(dWih, dWhh, decWp);
    pack_fc_kernel<<<(16 * 8 * 32 + 255) / 256, 256, 0, stream>>>(fcW, fcWp);
    init_state_kernel<<<(CBT * CH + 255) / 256, 256, 0, stream>>>(
        (unsigned short*)hb0, c_enc, c_dec, tok, sos);

    dim3 gGrid(CBT / 16, 4);   // 128 x 4 blocks, 4 waves/block, wave = one hidden-unit tile
    dim3 gBlk(128);

    // ---- encoder: 32 sequential fused LSTM steps, h ping-pong ----
    for (int t = 0; t < CTP; ++t) {
        const __bf16* hin = (t & 1) ? hb1 : hb0;
        __bf16* hout      = (t & 1) ? hb0 : hb1;
        lstm_step_fused<<<gGrid, gBlk, 0, stream>>>(
            phonemes, CTP, t, embp, hin, encWp, enc_b, hout, c_enc, enc_last, lens, t);
    }

    // ---- decoder: fused LSTM step + FC-to-d_out + argmax, h ping-pong with enc_last ----
    for (int t = 0; t < CTP; ++t) {
        const __bf16* hin = (t & 1) ? hd1 : enc_last;
        __bf16* hout      = (t & 1) ? enc_last : hd1;
        lstm_step_fused<<<gGrid, gBlk, 0, stream>>>(
            tok, 1, 0, embp, hin, decWp, dec_b, hout, c_dec, nullptr, nullptr, -1);
        fc_gemm<<<dim3(CBT / 16, 4), gBlk, 0, stream>>>(
            hout, fcWp, fcb, out + (size_t)t * CV, (long)CTP * CV);
        argmax_kernel<<<CBT * 32 / 256, 256, 0, stream>>>(
            out + (size_t)t * CV, (long)CTP * CV, tok);
    }
}
